// FacePartGAT_11751030522602
// MI455X (gfx1250) — compile-verified, matrix operand order checked by reading, not run on hardware
//
#include <hip/hip_runtime.h>

// ---------------------------------------------------------------------------
// CDNA5 (gfx1250) GAT pipeline: bf16 WMMA GEMMs + flash-style attention.
// wave32; WMMA shape 16x16x32 bf16 -> f32. Operands staged in LDS pre-packed
// in the ISA fragment layout so fragment builds are ds_load_b128 pairs.
// ---------------------------------------------------------------------------

typedef __attribute__((ext_vector_type(16))) __bf16 v16bf;
typedef __attribute__((ext_vector_type(2)))  __bf16 bf16x2;
typedef __attribute__((ext_vector_type(8)))  float  v8f;

union BfFrag { unsigned int u[8]; uint4 u4[2]; v16bf v; };

// f32x2 -> packed bf16x2 via hardware convert (RNE); lowers to v_cvt_*bf16_f32.
__device__ __forceinline__ unsigned int packbf(float a, float b) {
    union { bf16x2 h; unsigned int u; } cv;
    cv.h = bf16x2{(__bf16)a, (__bf16)b};
    return cv.u;
}
// K-slot base for 16-bit A/B fragment VGPR v (ISA 7.12.2, 16x32 bf16 layout).
__device__ __forceinline__ int kslot(int v, int half) {
    return (v < 4) ? (v * 2 + half * 8) : (16 + (v - 4) * 2 + half * 8);
}
__device__ __forceinline__ float lrelu(float x) { return x > 0.f ? x : 0.2f * x; }

// ---------------------------------------------------------------------------
// GEMM: C[MxN] = A[MxK] @ B[KxN], f32 in (bf16-packed in LDS), f32 out +
// bf16 copy packed in V-fragment layout: Cbp[row/32][col][kpair].
// Block = 256 threads = 8 wave32 (4x2), tile 128x128, BK=32, reg-pipelined.
// ---------------------------------------------------------------------------
__global__ __launch_bounds__(256) void gemm_bf16_kernel(
    const float* __restrict__ A, const float* __restrict__ B,
    float* __restrict__ C, unsigned int* __restrict__ Cbp,
    int M, int N, int K)
{
    __shared__ __align__(16) unsigned int As[128][16];   // [row][kpair]
    __shared__ __align__(16) unsigned int Bsp[128][16];  // [col][kpair]

    const int tid  = threadIdx.x;
    const int wave = tid >> 5, lane = tid & 31;
    const int row16 = lane & 15, half = lane >> 4;
    const int m0 = blockIdx.x * 128, n0 = blockIdx.y * 128;
    const int wm = wave >> 1, wn = wave & 1;  // wave tile: rows wm*32, cols wn*64

    v8f acc[2][4] = {};

    float4 aReg[4];
    float  bReg[16];

    auto loadGlobals = [&](int k0) {
#pragma unroll
        for (int i = 0; i < 4; ++i) {
            int idx = tid + i * 256;
            int r = idx >> 3, q = idx & 7;
            aReg[i] = *(const float4*)(A + (size_t)(m0 + r) * K + k0 + q * 4);
        }
#pragma unroll
        for (int i = 0; i < 8; ++i) {
            int idx = tid + i * 256;
            int col = idx & 127, kp = idx >> 7;
            bReg[2 * i + 0] = B[(size_t)(k0 + 2 * kp + 0) * N + n0 + col];
            bReg[2 * i + 1] = B[(size_t)(k0 + 2 * kp + 1) * N + n0 + col];
        }
    };
    auto storeLds = [&]() {
#pragma unroll
        for (int i = 0; i < 4; ++i) {
            int idx = tid + i * 256;
            int r = idx >> 3, q = idx & 7;
            As[r][q * 2 + 0] = packbf(aReg[i].x, aReg[i].y);
            As[r][q * 2 + 1] = packbf(aReg[i].z, aReg[i].w);
        }
#pragma unroll
        for (int i = 0; i < 8; ++i) {
            int idx = tid + i * 256;
            int col = idx & 127, kp = idx >> 7;
            Bsp[col][kp] = packbf(bReg[2 * i], bReg[2 * i + 1]);
        }
    };

    loadGlobals(0);
    storeLds();
    __syncthreads();

    for (int k0 = 0; k0 < K; k0 += 32) {
        const bool hasNext = (k0 + 32) < K;
        if (hasNext) loadGlobals(k0 + 32);   // global prefetch overlaps compute

        BfFrag afr[2], bfr[4];
#pragma unroll
        for (int mi = 0; mi < 2; ++mi) {
            int row = wm * 32 + mi * 16 + row16;
            const uint4* ap = (const uint4*)(&As[row][0]);
            afr[mi].u4[0] = ap[half];
            afr[mi].u4[1] = ap[2 + half];
        }
#pragma unroll
        for (int ni = 0; ni < 4; ++ni) {
            int col = wn * 64 + ni * 16 + row16;
            const uint4* bp = (const uint4*)(&Bsp[col][0]);
            bfr[ni].u4[0] = bp[half];
            bfr[ni].u4[1] = bp[2 + half];
        }
#pragma unroll
        for (int mi = 0; mi < 2; ++mi)
#pragma unroll
            for (int ni = 0; ni < 4; ++ni)
                acc[mi][ni] = __builtin_amdgcn_wmma_f32_16x16x32_bf16(
                    false, afr[mi].v, false, bfr[ni].v,
                    (short)0, acc[mi][ni], false, false);

        __syncthreads();
        if (hasNext) { storeLds(); __syncthreads(); }
    }

    // Epilogue. C/D layout: col = lane%16, row = vgpr + 8*(lane>=16).
    const int jb = (m0 + wm * 32) >> 5;   // 32-row block index
#pragma unroll
    for (int mi = 0; mi < 2; ++mi)
#pragma unroll
        for (int ni = 0; ni < 4; ++ni) {
            int col = n0 + wn * 64 + ni * 16 + row16;
#pragma unroll
            for (int r = 0; r < 8; ++r) {
                int row = m0 + wm * 32 + mi * 16 + r + 8 * half;
                C[(size_t)row * N + col] = acc[mi][ni][r];
            }
            // bf16 copy, packed: rows (2a, 2a+1 within tile) are vgprs 2a,2a+1.
#pragma unroll
            for (int a = 0; a < 4; ++a) {
                int kp = mi * 8 + 4 * half + a;
                Cbp[((size_t)jb * N + col) * 16 + kp] =
                    packbf(acc[mi][ni][2 * a], acc[mi][ni][2 * a + 1]);
            }
        }
}

// ---------------------------------------------------------------------------
// Attention scores: s[n,h] = <h[n,h,:], att_src[h,:]>, t likewise. C=128.
// ---------------------------------------------------------------------------
__global__ void scores_kernel(const float* __restrict__ h,
                              const float* __restrict__ asrc,
                              const float* __restrict__ adst,
                              float* __restrict__ s, float* __restrict__ t,
                              int HC, int H)
{
    int n = blockIdx.x;
    int head = threadIdx.x >> 5, lane = threadIdx.x & 31;
    float ss = 0.f, tt = 0.f;
#pragma unroll
    for (int c = lane; c < 128; c += 32) {
        float hv = h[(size_t)n * HC + head * 128 + c];
        ss += hv * asrc[head * 128 + c];
        tt += hv * adst[head * 128 + c];
    }
#pragma unroll
    for (int off = 16; off > 0; off >>= 1) {
        ss += __shfl_down(ss, off, 32);
        tt += __shfl_down(tt, off, 32);
    }
    if (lane == 0) { s[n * H + head] = ss; t[n * H + head] = tt; }
}

// ---------------------------------------------------------------------------
// Flash attention aggregation over a fully connected graph.
// V operand arrives pre-packed: Vp[j/32][col (HC wide)][kpair] (uint = 2 bf16).
// Block = 256 threads (8 waves), 128 dest rows/block (16 per wave), C = 128.
// ---------------------------------------------------------------------------
__global__ __launch_bounds__(256) void flash_attn_kernel(
    const unsigned int* __restrict__ Vp,
    const float* __restrict__ s, const float* __restrict__ t,  // [N, H]
    float* __restrict__ Out,                                   // f32 [N, HC]
    int N, int HC, int H)
{
    __shared__ __align__(16) unsigned int Hp[128][16];  // [col][kpair]
    __shared__ float Ssh[32];

    const int head = blockIdx.y;
    const int coff = head * 128;
    const int tid  = threadIdx.x;
    const int wave = tid >> 5, lane = tid & 31;
    const int row16 = lane & 15, half = lane >> 4;
    const int rbase = blockIdx.x * 128 + wave * 16;
    const float trow = t[(size_t)(rbase + row16) * H + head];

    v8f acc[8] = {};
    float m_i = -1e30f, l_i = 0.f;

    uint4 vReg[2];
    float sReg = 0.f;

    auto loadGlobals = [&](int j0) {
        const uint4* src =
            (const uint4*)(Vp + ((size_t)(j0 >> 5) * HC + coff) * 16) + tid * 2;
        vReg[0] = src[0];
        vReg[1] = src[1];
        if (tid < 32) sReg = s[(size_t)(j0 + tid) * H + head];
    };
    auto storeLds = [&]() {
        uint4* dst = (uint4*)(&Hp[0][0]) + tid * 2;
        dst[0] = vReg[0];
        dst[1] = vReg[1];
        if (tid < 32) Ssh[tid] = sReg;
    };

    loadGlobals(0);
    storeLds();
    __syncthreads();

    for (int j0 = 0; j0 < N; j0 += 32) {
        const bool hasNext = (j0 + 32) < N;
        if (hasNext) loadGlobals(j0 + 32);

        // Logits for this lane's 16 K slots (A-fragment layout).
        float e0[8], e1[8], mymax = -1e30f;
#pragma unroll
        for (int v = 0; v < 8; ++v) {
            int kb = kslot(v, half);
            e0[v] = lrelu(trow + Ssh[kb]);
            e1[v] = lrelu(trow + Ssh[kb + 1]);
            mymax = fmaxf(mymax, fmaxf(e0[v], e1[v]));
        }
        float rowmax = fmaxf(mymax, __shfl_xor(mymax, 16, 32));
        float m_new  = fmaxf(m_i, rowmax);
        float scale  = __expf(m_i - m_new);

        float rsum = 0.f;
        BfFrag pf;
#pragma unroll
        for (int v = 0; v < 8; ++v) {
            float p0 = __expf(e0[v] - m_new);
            float p1 = __expf(e1[v] - m_new);
            rsum += p0 + p1;
            pf.u[v] = packbf(p0, p1);
        }
        rsum += __shfl_xor(rsum, 16, 32);
        l_i = l_i * scale + rsum;
        m_i = m_new;

        // Rescale accumulators (factor for accum row r+8*half lives in that lane).
        float sc[8];
#pragma unroll
        for (int r = 0; r < 8; ++r) sc[r] = __shfl(scale, r + 8 * half, 32);
#pragma unroll
        for (int ni = 0; ni < 8; ++ni)
#pragma unroll
            for (int r = 0; r < 8; ++r) acc[ni][r] *= sc[r];

        // P @ V: 8 WMMAs over the 128 output columns; B frags via ds_load_b128.
#pragma unroll
        for (int ni = 0; ni < 8; ++ni) {
            BfFrag bf;
            int col = ni * 16 + row16;
            const uint4* bp = (const uint4*)(&Hp[col][0]);
            bf.u4[0] = bp[half];
            bf.u4[1] = bp[2 + half];
            acc[ni] = __builtin_amdgcn_wmma_f32_16x16x32_bf16(
                false, pf.v, false, bf.v, (short)0, acc[ni], false, false);
        }

        __syncthreads();
        if (hasNext) { storeLds(); __syncthreads(); }
    }

    // Finalize: divide by softmax denominator and store f32.
    float linv[8];
#pragma unroll
    for (int r = 0; r < 8; ++r)
        linv[r] = 1.f / __shfl(l_i, r + 8 * half, 32);
#pragma unroll
    for (int ni = 0; ni < 8; ++ni)
#pragma unroll
        for (int r = 0; r < 8; ++r) {
            int row = rbase + r + 8 * half;
            int col = coff + ni * 16 + row16;
            Out[(size_t)row * HC + col] = acc[ni][r] * linv[r];
        }
}

// ---------------------------------------------------------------------------
// x = elu(x + bias[col]), in place. ncol must be a power of two.
// ---------------------------------------------------------------------------
__global__ void elu_bias_kernel(float* __restrict__ x, const float* __restrict__ b,
                                int total, int ncolMask)
{
    int i = blockIdx.x * blockDim.x + threadIdx.x;
    if (i < total) {
        float v = x[i] + b[i & ncolMask];
        x[i] = v > 0.f ? v : (__expf(v) - 1.f);
    }
}

// Column mean over N rows.
__global__ void colmean_kernel(const float* __restrict__ y, float* __restrict__ m,
                               int N, int C)
{
    __shared__ float red[256];
    int col = blockIdx.x;
    float sum = 0.f;
    for (int n = threadIdx.x; n < N; n += 256) sum += y[(size_t)n * C + col];
    red[threadIdx.x] = sum;
    __syncthreads();
    for (int off = 128; off > 0; off >>= 1) {
        if (threadIdx.x < off) red[threadIdx.x] += red[threadIdx.x + off];
        __syncthreads();
    }
    if (threadIdx.x == 0) m[col] = red[0] / (float)N;
}

// out[d] = sum_c mean[c] * fcW[c, d] + fcb[d]
__global__ void final_kernel(const float* __restrict__ mean,
                             const float* __restrict__ fcW,
                             const float* __restrict__ fcb,
                             float* __restrict__ out, int C, int D)
{
    int d = blockIdx.x * blockDim.x + threadIdx.x;
    if (d < D) {
        float a = fcb[d];
        for (int c = 0; c < C; ++c) a += mean[c] * fcW[(size_t)c * D + d];
        out[d] = a;
    }
}

// ---------------------------------------------------------------------------
extern "C" void kernel_launch(void* const* d_in, const int* in_sizes, int n_in,
                              void* d_out, int out_size, void* d_ws, size_t ws_size,
                              hipStream_t stream)
{
    const int N = 4096, D = 768, C = 128, H = 4, HC = H * C;  // 512

    const float* X    = (const float*)d_in[0];
    const float* W1   = (const float*)d_in[1];
    const float* as1  = (const float*)d_in[2];
    const float* ad1  = (const float*)d_in[3];
    const float* b1   = (const float*)d_in[4];
    const float* W2   = (const float*)d_in[5];
    const float* as2  = (const float*)d_in[6];
    const float* ad2  = (const float*)d_in[7];
    const float* b2   = (const float*)d_in[8];
    const float* fcW  = (const float*)d_in[9];
    const float* fcb  = (const float*)d_in[10];
    float* out = (float*)d_out;

    char* ws = (char*)d_ws;
    size_t off = 0;
    auto alloc = [&](size_t bytes) { char* p = ws + off; off += (bytes + 255) & ~(size_t)255; return p; };

    float*        h1f = (float*)alloc((size_t)N * HC * 4);
    unsigned int* h1p = (unsigned int*)alloc((size_t)N * HC * 2);  // packed bf16
    float*        s1  = (float*)alloc((size_t)N * H * 4);
    float*        t1  = (float*)alloc((size_t)N * H * 4);
    float*        x2  = (float*)alloc((size_t)N * HC * 4);         // gat1 out -> elu in place
    float*        h2f = (float*)alloc((size_t)N * C * 4);
    unsigned int* h2p = (unsigned int*)alloc((size_t)N * C * 2);   // packed bf16
    float*        s2  = (float*)alloc((size_t)N * 4);
    float*        t2  = (float*)alloc((size_t)N * 4);
    float*        y2  = (float*)alloc((size_t)N * C * 4);          // gat2 out -> elu in place
    float*        mv  = (float*)alloc((size_t)C * 4);
    (void)ws_size; (void)in_sizes; (void)n_in; (void)out_size;

    // Layer 1: h1 = X @ W1   (4096 x 768 x 512)
    gemm_bf16_kernel<<<dim3(N / 128, HC / 128), 256, 0, stream>>>(X, W1, h1f, h1p, N, HC, D);
    scores_kernel<<<N, 32 * H, 0, stream>>>(h1f, as1, ad1, s1, t1, HC, H);
    flash_attn_kernel<<<dim3(N / 128, H), 256, 0, stream>>>(h1p, s1, t1, x2, N, HC, H);
    elu_bias_kernel<<<(N * HC) / 256, 256, 0, stream>>>(x2, b1, N * HC, HC - 1);

    // Layer 2: h2 = x2 @ W2  (4096 x 512 x 128), single head
    gemm_bf16_kernel<<<dim3(N / 128, C / 128), 256, 0, stream>>>(x2, W2, h2f, h2p, N, C, HC);
    scores_kernel<<<N, 32, 0, stream>>>(h2f, as2, ad2, s2, t2, C, 1);
    flash_attn_kernel<<<dim3(N / 128, 1), 256, 0, stream>>>(h2p, s2, t2, y2, N, C, 1);
    elu_bias_kernel<<<(N * C) / 256, 256, 0, stream>>>(y2, b2, N * C, C - 1);

    // Mean over nodes, then final linear 128 -> 768.
    colmean_kernel<<<C, 256, 0, stream>>>(y2, mv, N, C);
    final_kernel<<<(D + 255) / 256, 256, 0, stream>>>(mv, fcW, fcb, out, C, D);
}
